// AnalogConv2d_20718922236507
// MI455X (gfx1250) — matrix operand order repaired and checked by reading, not compiled, not run on hardware
//
#include <hip/hip_runtime.h>

typedef __attribute__((ext_vector_type(16))) _Float16 v16h;
typedef __attribute__((ext_vector_type(8)))  _Float16 v8h;   // 16 B = one ds_load_b128
typedef __attribute__((ext_vector_type(8)))  float    v8f;

#define C_IN   128
#define C_OUT  256
#define HW     56
#define L_IMG  (56 * 56)
#define NTILE  32              // output positions per workgroup
#define NCB    4               // 128 channels / 32 per K-block
#define RPITCH 28              // dwords per (r,wp) row: 16 used + 12 pad (bank spread)
#define NBIDX  (4 * 58)        // r in 0..3, wp in 0..57
#define LO_OFS (NBIDX * RPITCH)
#define TILE_F16 1024          // f16 elems per packed A tile (32 lanes * 16 * {hi,lo})

// ---------------------------------------------------------------------------
// Repack W[256,128,3,3] fp32 into per-WMMA-tile A fragments (hi plane, lo plane),
// in contiguous-k lane order (lane half h holds k 16h..16h+15): fragment loads become contiguous b128s.
// tile id = (mtile*9 + kidx)*4 + cb.
// ---------------------------------------------------------------------------
__global__ __launch_bounds__(256) void pack_weights_kernel(const float* __restrict__ w,
                                                           _Float16* __restrict__ wpk) {
    int t    = blockIdx.x * blockDim.x + threadIdx.x;   // 576 tiles * 32 lanes = 18432
    int lane = t & 31;
    int tile = t >> 5;                                   // 0..575
    int cb   = tile & 3;
    int kidx = (tile >> 2) % 9;
    int mt   = tile / 36;                                // 0..15
    int m     = mt * 16 + (lane & 15);
    int khalf = (lane >> 4) * 8;                         // lanes16-31 hold K 16..31
    int ki = kidx / 3, kj = kidx % 3;
    v16h hi, lo;
#pragma unroll
    for (int e = 0; e < 16; ++e) {
        int k32 = 2 * khalf + e;
        int c   = cb * 32 + k32;
        float v = w[((m * C_IN + c) * 3 + ki) * 3 + kj];
        _Float16 h = (_Float16)v;
        hi[e] = h;
        lo[e] = (_Float16)(v - (float)h);
    }
    _Float16* base = wpk + (size_t)tile * TILE_F16;
    *(v16h*)(base + lane * 16)       = hi;
    *(v16h*)(base + 512 + lane * 16) = lo;
}

// One 32(K) x 16(N) B-fragment plane: the lane's 8 channel-pair dwords are
// CONTIGUOUS in LDS -> two ds_load_b128 written straight into the two halves
// of the v16h operand tuple (no repacking moves).
static __device__ __forceinline__ v16h load_frag(const unsigned* __restrict__ sm,
                                                 int dwbase) {
    union { v16h h; v8h q[2]; } r;
    r.q[0] = *(const v8h*)(sm + dwbase);       // dwords 0..3  (16B aligned)
    r.q[1] = *(const v8h*)(sm + dwbase + 4);   // dwords 4..7  (16B aligned)
    return r.h;
}

// A-fragment plane at a compile-time byte offset from a hoisted base pointer,
// so the global_load_b128s use constant instruction offsets (no per-kidx
// 64-bit address VALU).
static __device__ __forceinline__ v16h load_afrag(const _Float16* __restrict__ base,
                                                  int f16_ofs) {
    union { v16h h; v8h q[2]; } r;
    r.q[0] = *(const v8h*)(base + f16_ofs);
    r.q[1] = *(const v8h*)(base + f16_ofs + 8);
    return r.h;
}

__global__ __launch_bounds__(256) void conv3x3_wmma_kernel(const float* __restrict__ x,
                                                           const _Float16* __restrict__ wpk,
                                                           const float* __restrict__ bias,
                                                           float* __restrict__ out) {
    // (r,wp)-major: row = 16 channel-pair dwords (+12 pad); hi plane then lo plane.
    __shared__ __align__(16) unsigned smem[2 * NBIDX * RPITCH];   // 51968 B

    const int tid  = threadIdx.x;
    const int lane = tid & 31;
    const int wv   = tid >> 5;               // 0..7 waves
    const int lBase = blockIdx.x * NTILE;    // 98 tiles * 32 = 3136 = 56*56 exact
    const int n     = blockIdx.y;            // batch
    const int mblk  = blockIdx.z;            // 0..1 -> 128 Cout rows each
    const int h_first = lBase / HW;
    const int hTop    = h_first - 1;         // staged rows hTop..hTop+3 cover kh = 0..2

    const int col  = lane & 15;
    const int hi16 = lane >> 4;
    const int pairBase = hi16 * 8;           // lanes16-31 read channels 16..31

    // spatial decode for the two 16-wide N subtiles (tile spans at most 2 image rows)
    const int l0 = lBase + col;
    const int l1 = lBase + 16 + col;
    const int dr0 = l0 / HW - h_first, w0 = l0 % HW;
    const int dr1 = l1 / HW - h_first, w1 = l1 % HW;

    const int mtile = mblk * 8 + wv;         // global 16-row Cout tile

    // Hoisted A base: per-kidx/per-cb displacement is a compile-time constant
    // (cb stride = TILE_F16, kidx stride = 4*TILE_F16 f16 elems).
    const _Float16* aw = wpk + (size_t)(mtile * 9 * 4) * TILE_F16 + lane * 16;

    v8f acc0 = {};
    v8f acc1 = {};

    for (int cb = 0; cb < NCB; ++cb) {
        __syncthreads();
        // Stage x[n, cb*32..+32, hTop..hTop+3, -1..56]: split each f32 into {hi,lo}
        // f16 once, packed as channel-PAIR dwords, (r,wp)-major so reads are b128s.
        for (int s = tid; s < 16 * 4 * 58; s += 256) {
            int wp_ = s % 58;
            int pr  = s / 58;
            int r   = pr & 3;
            int cp  = pr >> 2;               // channel pair 0..15
            int h_in = hTop + r;
            int w_in = wp_ - 1;
            float v0 = 0.0f, v1 = 0.0f;
            if ((unsigned)h_in < (unsigned)HW && (unsigned)w_in < (unsigned)HW) {
                const float* px =
                    x + (((size_t)n * C_IN + cb * 32 + 2 * cp) * HW + h_in) * HW + w_in;
                v0 = px[0];
                v1 = px[L_IMG];              // next channel, same (h,w)
            }
            union { _Float16 f[2]; unsigned u; } hw, lw;
            _Float16 h0 = (_Float16)v0, h1 = (_Float16)v1;
            hw.f[0] = h0;                     hw.f[1] = h1;
            lw.f[0] = (_Float16)(v0 - (float)h0);
            lw.f[1] = (_Float16)(v1 - (float)h1);
            int row = (r * 58 + wp_) * RPITCH;
            smem[row + cp]          = hw.u;
            smem[LO_OFS + row + cp] = lw.u;
        }
        __syncthreads();

        if (cb < NCB - 1) {   // warm next channel block toward L2/WGP$
            const char* pf = (const char*)(x + (((size_t)n * C_IN + (cb + 1) * 32) * HW) * HW);
            __builtin_prefetch(pf + tid * 256, 0, 0);
        }

        const _Float16* acb = aw + cb * TILE_F16;

#pragma unroll
        for (int kidx = 0; kidx < 9; ++kidx) {
            const int ki = kidx / 3, kj = kidx % 3;
            const int aofs = kidx * 4 * TILE_F16;        // compile-time constant

            v16h ahi = load_afrag(acb, aofs);
            v16h alo = load_afrag(acb, aofs + 512);

            const int d0 = ((dr0 + ki) * 58 + (w0 + kj)) * RPITCH + pairBase;
            const int d1 = ((dr1 + ki) * 58 + (w1 + kj)) * RPITCH + pairBase;

            v16h b0h = load_frag(smem, d0);
            v16h b0l = load_frag(smem + LO_OFS, d0);
            v16h b1h = load_frag(smem, d1);
            v16h b1l = load_frag(smem + LO_OFS, d1);

            // Interleave the two accumulators so consecutive WMMAs are independent.
            acc0 = __builtin_amdgcn_wmma_f32_16x16x32_f16(false, ahi, false, b0h,
                                                          (short)0, acc0, false, false);
            acc1 = __builtin_amdgcn_wmma_f32_16x16x32_f16(false, ahi, false, b1h,
                                                          (short)0, acc1, false, false);
            acc0 = __builtin_amdgcn_wmma_f32_16x16x32_f16(false, ahi, false, b0l,
                                                          (short)0, acc0, false, false);
            acc1 = __builtin_amdgcn_wmma_f32_16x16x32_f16(false, ahi, false, b1l,
                                                          (short)0, acc1, false, false);
            acc0 = __builtin_amdgcn_wmma_f32_16x16x32_f16(false, alo, false, b0h,
                                                          (short)0, acc0, false, false);
            acc1 = __builtin_amdgcn_wmma_f32_16x16x32_f16(false, alo, false, b1h,
                                                          (short)0, acc1, false, false);
        }
    }

    // Epilogue: C/D layout -> VGPR r holds M = r + 8*hi16; lane&15 is N column.
    const int oBase = mtile * 16 + hi16 * 8;
#pragma unroll
    for (int r = 0; r < 8; ++r) {
        int o   = oBase + r;
        float b = bias[o];
        out[((size_t)n * C_OUT + o) * L_IMG + l0] = acc0[r] + b;
        out[((size_t)n * C_OUT + o) * L_IMG + l1] = acc1[r] + b;
    }
}

extern "C" void kernel_launch(void* const* d_in, const int* in_sizes, int n_in,
                              void* d_out, int out_size, void* d_ws, size_t ws_size,
                              hipStream_t stream) {
    const float* x    = (const float*)d_in[0];
    const float* w    = (const float*)d_in[1];
    const float* bias = (const float*)d_in[2];
    float* out        = (float*)d_out;

    // Packed weights: 576 tiles * 1024 f16 = 1,179,648 B of workspace.
    _Float16* wpk = (_Float16*)d_ws;

    pack_weights_kernel<<<72, 256, 0, stream>>>(w, wpk);

    dim3 grid(98 /* N tiles */, 16 /* batch */, 2 /* Cout/128 */);
    conv3x3_wmma_kernel<<<grid, 256, 0, stream>>>(x, wpk, bias, out);
}
